// IntraAttention_85289460564635
// MI455X (gfx1250) — matrix operand order repaired and checked by reading, not compiled, not run on hardware
//
#include <hip/hip_runtime.h>
#include <math.h>

// Problem constants (from reference): B=4, S=2048, D_IN=D_OUT=H=1024
#define NBATCH 4
#define SEQ    2048
#define HDIM   1024
#define MAXD   10
#define BSCALE 0.01f

typedef __attribute__((ext_vector_type(16))) __bf16 v16bf;
typedef __attribute__((ext_vector_type(8)))  float  v8f;

union Frag {            // one WMMA 16-bit A/B fragment: 16 bf16 per lane = 32B
    v16bf bf;
    uint4 u[2];
};

__device__ __forceinline__ __bf16 f2bf(float f) {
    unsigned int u = __builtin_bit_cast(unsigned int, f);
    u += 0x7FFFu + ((u >> 16) & 1u);          // round-to-nearest-even
    unsigned short h = (unsigned short)(u >> 16);
    return __builtin_bit_cast(__bf16, h);
}

// ---------------------------------------------------------------------------
// Kernel 0: bulk f32 -> bf16 conversion (bandwidth-bound, one pass).
// ---------------------------------------------------------------------------
__global__ __launch_bounds__(256) void cvt_kernel(
    const float* __restrict__ src, __bf16* __restrict__ dst, int n4)
{
    const int i = blockIdx.x * 256 + threadIdx.x;
    if (i >= n4) return;
    const float4 v = ((const float4*)src)[i];
    union { __bf16 b[4]; short4 s; } o;
    o.b[0] = f2bf(v.x); o.b[1] = f2bf(v.y);
    o.b[2] = f2bf(v.z); o.b[3] = f2bf(v.w);
    ((short4*)dst)[i] = o.s;
}

// ---------------------------------------------------------------------------
// Kernel 1: f = x @ W^T + b   (M = B*S = 8192, N = H = 1024, K = 1024)
// Each wave computes a 16x128 strip (8 tiles). Fragments are double-buffered
// in registers so loads for step n+1 are in flight while WMMA n executes.
// Writes f as bf16 in two layouts:
//   fR[b][s][h]  (row-major: score-A rows & score-B columns, contiguous in h)
//   fT[b][h][s]  (transposed: PV-B columns, contiguous in s)
// ---------------------------------------------------------------------------
__global__ __launch_bounds__(256) void linear_bf16_kernel(
    const __bf16* __restrict__ xb, const __bf16* __restrict__ Wb,
    const float* __restrict__ bias,
    __bf16* __restrict__ fR, __bf16* __restrict__ fT)
{
    const int lane = threadIdx.x & 31;
    const int wave = threadIdx.x >> 5;
    const int half = lane >> 4;            // 0: lanes 0-15, 1: lanes 16-31
    const int l16  = lane & 15;

    const int rowbase = blockIdx.x * 16;   // over B*S
    const int colbase = wave * 128;        // over H (8 waves cover all 1024)

    const __bf16* arow = xb + (size_t)(rowbase + l16) * HDIM;           // A row = l16
    const __bf16* bcol = Wb + (size_t)(colbase + l16) * HDIM + half * 16; // B col = l16

    v8f acc[8];
    {
        v8f z = {};
        #pragma unroll
        for (int t = 0; t < 8; ++t) acc[t] = z;
    }

    // A k order chosen here (same as B): half0 -> K kb..kb+15, half1 -> K kb+16..kb+31
    Frag fa;
    fa.u[0] = ((const uint4*)(arow + half * 16))[0];
    fa.u[1] = ((const uint4*)(arow + half * 16 + 8))[0];
    Frag fa_n = fa;

    for (int kb = 0; kb < HDIM; kb += 32) {
        Frag fg, fg_n;
        // B k order chosen here (same as A): half0 -> K kb..kb+15, half1 -> K kb+16..kb+31
        fg.u[0] = ((const uint4*)(bcol + kb))[0];
        fg.u[1] = ((const uint4*)(bcol + kb))[1];
        fg_n = fg;
        if (kb + 32 < HDIM) {              // prefetch next A fragment
            fa_n.u[0] = ((const uint4*)(arow + kb + 32 + half * 16))[0];
            fa_n.u[1] = ((const uint4*)(arow + kb + 40 + half * 16))[0];
        }
        #pragma unroll
        for (int t = 0; t < 8; ++t) {
            if (t < 7) {                   // prefetch next tile's B fragment
                const uint4* bp = (const uint4*)(bcol + (size_t)(t + 1) * 16 * HDIM + kb);
                fg_n.u[0] = bp[0];
                fg_n.u[1] = bp[1];
            }
            acc[t] = __builtin_amdgcn_wmma_f32_16x16x32_bf16(
                         false, fa.bf, false, fg.bf, (short)0, acc[t], false, false);
            fg = fg_n;
        }
        fa = fa_n;
    }

    // C layout: lanes 0-15: VGPR r -> M=r; lanes 16-31: VGPR r -> M=r+8; N=l16
    const int bb = rowbase >> 11;                       // / SEQ (16 divides SEQ)
    const int sbase = rowbase & (SEQ - 1);
    #pragma unroll
    for (int t = 0; t < 8; ++t) {
        const int col = colbase + t * 16 + l16;
        const float bv = bias[col];
        #pragma unroll
        for (int r = 0; r < 8; ++r) {
            const int row = rowbase + r + half * 8;     // global flat row
            const __bf16 h = f2bf(acc[t][r] + bv);
            fR[(size_t)row * HDIM + col] = h;
            fT[((size_t)bb * HDIM + col) * SEQ + (sbase + r + half * 8)] = h;
        }
    }
}

// ---------------------------------------------------------------------------
// Kernel 2: flash attention over f with distance bias.
// Block = 256 threads (8 waves) per (batch, 16-row) slab.
// Wave w owns score columns [jb+16w, +16), softmax rows {2w, 2w+1}, and
// output columns [128w, 128w+128).
// ---------------------------------------------------------------------------
__global__ __launch_bounds__(256) void attn_kernel(
    const __bf16* __restrict__ fR, const __bf16* __restrict__ fT,
    float* __restrict__ out)
{
    __shared__ __bf16 fi_lds[16][HDIM];   // 32 KB: the 16 query rows, bf16
    __shared__ float  s_lds[16][128];     //  8 KB: raw scores for current j-block
    __shared__ __bf16 p_lds[16][128];     //  4 KB: softmax numerators (bf16)
    __shared__ float  c_lds[16];          // per-row rescale factor
    __shared__ float  l_lds[16];          // final per-row denominators

    const int tid  = threadIdx.x;
    const int lane = tid & 31;
    const int wave = tid >> 5;
    const int half = lane >> 4;
    const int l16  = lane & 15;

    const int b     = blockIdx.y;
    const int ibase = blockIdx.x * 16;
    const __bf16* fb  = fR + (size_t)b * SEQ * HDIM;
    const __bf16* fTb = fT + (size_t)b * HDIM * SEQ;

    // Stage f_i (16 x 1024 bf16) into LDS: thread t copies 128B of row t>>4.
    {
        const int r = tid >> 4, seg = tid & 15;
        const uint4* src = (const uint4*)(fb + (size_t)(ibase + r) * HDIM) + seg * 8;
        uint4*       dst = (uint4*)(&fi_lds[r][0]) + seg * 8;
        #pragma unroll
        for (int i = 0; i < 8; ++i) dst[i] = src[i];
    }
    __syncthreads();

    v8f acc[8];
    {
        v8f z = {};
        #pragma unroll
        for (int t = 0; t < 8; ++t) acc[t] = z;
    }

    // Online-softmax state: lane tracks row srow (replicated across its
    // 16-lane half); each lane owns 8 contiguous columns of that row.
    const int srow  = 2 * wave + half;
    const int cbase = l16 * 8;
    float m_run = -__builtin_inff();
    float l_run = 0.0f;

    for (int jb = 0; jb < SEQ; jb += 128) {
        // ---- 1) scores: wave w -> 16x16 tile at local cols [16w,16w+16)
        v8f sc = {};
        const __bf16* fj = fb + (size_t)(jb + wave * 16 + l16) * HDIM; // B col = key row
        if (jb + 128 < SEQ)
            __builtin_prefetch(fj + 128 * HDIM, 0, 1);                 // next j-block

        Frag fa, fg;
        fa.u[0] = ((const uint4*)(&fi_lds[l16][half * 16]))[0];
        fa.u[1] = ((const uint4*)(&fi_lds[l16][half * 16 + 8]))[0];
        {
            const uint4* bp = (const uint4*)(fj + half * 16);
            fg.u[0] = bp[0];
            fg.u[1] = bp[1];
        }
        for (int kb = 0; kb < HDIM; kb += 32) {
            Frag fa_n = fa, fg_n = fg;
            if (kb + 32 < HDIM) {          // prefetch next step's fragments
                fa_n.u[0] = ((const uint4*)(&fi_lds[l16][kb + 32 + half * 16]))[0];
                fa_n.u[1] = ((const uint4*)(&fi_lds[l16][kb + 40 + half * 16]))[0];
                const uint4* bp = (const uint4*)(fj + kb + 32 + half * 16);
                fg_n.u[0] = bp[0];
                fg_n.u[1] = bp[1];
            }
            sc = __builtin_amdgcn_wmma_f32_16x16x32_bf16(
                     false, fa.bf, false, fg.bf, (short)0, sc, false, false);
            fa = fa_n;
            fg = fg_n;
        }
        #pragma unroll
        for (int r = 0; r < 8; ++r) {
            const int irow = r + half * 8;          // local query row 0..15
            const int jcol = wave * 16 + l16;       // local key col 0..127
            int d = (ibase + irow) - (jb + jcol);
            d = d < 0 ? -d : d;
            d = d > MAXD ? MAXD : d;
            s_lds[irow][jcol] = sc[r] + BSCALE * (float)d;
        }
        __syncthreads();

        // ---- 2) online softmax: all 8 waves, 2 rows per wave, 8 cols per lane
        {
            float sv[8];
            float mb = -__builtin_inff();
            #pragma unroll
            for (int j = 0; j < 8; ++j) {
                sv[j] = s_lds[srow][cbase + j];
                mb = fmaxf(mb, sv[j]);
            }
            #pragma unroll
            for (int m = 1; m < 16; m <<= 1)         // stays within 16-lane half
                mb = fmaxf(mb, __shfl_xor(mb, m, 32));
            const float m_new = fmaxf(m_run, mb);
            const float cf    = __expf(m_run - m_new);
            float ls = 0.0f;
            #pragma unroll
            for (int j = 0; j < 8; ++j) {
                const float p = __expf(sv[j] - m_new);
                ls += p;
                p_lds[srow][cbase + j] = f2bf(p);
            }
            #pragma unroll
            for (int m = 1; m < 16; m <<= 1)
                ls += __shfl_xor(ls, m, 32);
            l_run = l_run * cf + ls;
            m_run = m_new;
            if (l16 == 0) c_lds[srow] = cf;
        }
        __syncthreads();

        // ---- 3) rescale accumulators, then acc += P (16x128) x V (128x16)
        float cf[8];
        #pragma unroll
        for (int r = 0; r < 8; ++r) cf[r] = c_lds[r + half * 8];
        #pragma unroll
        for (int t = 0; t < 8; ++t) {
            #pragma unroll
            for (int r = 0; r < 8; ++r) acc[t][r] = acc[t][r] * cf[r];
        }
        #pragma unroll
        for (int kb = 0; kb < 128; kb += 32) {
            Frag pa;
            pa.u[0] = ((const uint4*)(&p_lds[l16][kb + half * 16]))[0];
            pa.u[1] = ((const uint4*)(&p_lds[l16][kb + half * 16 + 8]))[0];
            const __bf16* vcol =
                fTb + (size_t)(wave * 128 + l16) * SEQ + jb + kb + half * 16;
            Frag fg2, fg2_n;
            fg2.u[0] = ((const uint4*)vcol)[0];
            fg2.u[1] = ((const uint4*)vcol)[1];
            fg2_n = fg2;
            #pragma unroll
            for (int t = 0; t < 8; ++t) {
                if (t < 7) {               // prefetch next tile's V fragment
                    const uint4* bp = (const uint4*)(vcol + (size_t)(t + 1) * 16 * SEQ);
                    fg2_n.u[0] = bp[0];
                    fg2_n.u[1] = bp[1];
                }
                acc[t] = __builtin_amdgcn_wmma_f32_16x16x32_bf16(
                             false, pa.bf, false, fg2.bf, (short)0, acc[t], false, false);
                fg2 = fg2_n;
            }
        }
        // No extra barrier needed: next iter's s_lds writes are fenced by the
        // step-1 barrier before any wave touches p_lds/c_lds again.
    }

    if (l16 == 0) l_lds[srow] = l_run;
    __syncthreads();

    float linv[8];
    #pragma unroll
    for (int r = 0; r < 8; ++r) linv[r] = 1.0f / l_lds[r + half * 8];
    float* outb = out + ((size_t)b * SEQ + ibase) * HDIM;
    #pragma unroll
    for (int t = 0; t < 8; ++t) {
        const int hcol = wave * 128 + t * 16 + l16;
        #pragma unroll
        for (int r = 0; r < 8; ++r) {
            const int irow = r + half * 8;
            outb[(size_t)irow * HDIM + hcol] = acc[t][r] * linv[r];
        }
    }
}

// ---------------------------------------------------------------------------
// Launch. ws layout (bf16):
//   [ xb : 16MB ][ Wb : 2MB ][ fR : 16MB ][ fT : 16MB ]   (needs 50MB ws)
// ---------------------------------------------------------------------------
extern "C" void kernel_launch(void* const* d_in, const int* in_sizes, int n_in,
                              void* d_out, int out_size, void* d_ws, size_t ws_size,
                              hipStream_t stream) {
    (void)in_sizes; (void)n_in; (void)out_size; (void)ws_size;
    const float* x = (const float*)d_in[0];
    const float* W = (const float*)d_in[1];
    const float* b = (const float*)d_in[2];
    float* out = (float*)d_out;

    __bf16* xb = (__bf16*)d_ws;
    __bf16* Wb = xb + (size_t)NBATCH * SEQ * HDIM;
    __bf16* fR = Wb + (size_t)HDIM * HDIM;
    __bf16* fT = fR + (size_t)NBATCH * SEQ * HDIM;

    dim3 blk(256);

    const int nx4 = (NBATCH * SEQ * HDIM) / 4;          // 2M float4 groups
    const int nw4 = (HDIM * HDIM) / 4;                  // 256K float4 groups
    hipLaunchKernelGGL(cvt_kernel, dim3((nx4 + 255) / 256), blk, 0, stream, x, xb, nx4);
    hipLaunchKernelGGL(cvt_kernel, dim3((nw4 + 255) / 256), blk, 0, stream, W, Wb, nw4);

    dim3 g1((NBATCH * SEQ) / 16);                       // 512 workgroups
    hipLaunchKernelGGL(linear_bf16_kernel, g1, blk, 0, stream, xb, Wb, b, fR, fT);

    dim3 g2(SEQ / 16, NBATCH);                          // 128 x 4 workgroups
    hipLaunchKernelGGL(attn_kernel, g2, blk, 0, stream, fR, fT, out);
}
